// Decoder_15496242004095
// MI455X (gfx1250) — compile-verified, matrix operand order checked by reading, not run on hardware
//
#include <hip/hip_runtime.h>
#include <hip/hip_bf16.h>
#include <math.h>

typedef _Float16 f16;
typedef __attribute__((ext_vector_type(16))) _Float16 v16h;
typedef __attribute__((ext_vector_type(8)))  _Float16 v8h;
typedef __attribute__((ext_vector_type(8)))  float    v8f;
typedef unsigned int u32x4 __attribute__((ext_vector_type(4)));
typedef int          i32x4 __attribute__((ext_vector_type(4)));
typedef int          i32x8 __attribute__((ext_vector_type(8)));

#ifndef __has_builtin
#define __has_builtin(x) 0
#endif
#if __has_builtin(__builtin_amdgcn_tensor_load_to_lds) && \
    __has_builtin(__builtin_amdgcn_s_wait_tensorcnt)
#define USE_TDM 1
#else
#define USE_TDM 0
#endif

// ---------------------------------------------------------------------------
// WMMA helpers (CDNA5 / gfx1250, wave32). D = A(16x32) * B(32x16) + C, f32 acc
// ---------------------------------------------------------------------------
__device__ __forceinline__ v8f wmma16(v16h a, v16h b, v8f c) {
  return __builtin_amdgcn_wmma_f32_16x16x32_f16(
      /*neg_a=*/false, a, /*neg_b=*/false, b,
      /*c_mod=*/(short)0, c, /*reuse_a=*/false, /*reuse_b=*/false);
}

// A fragment: 16x32 f16 tile from a [16 x rowStride] f16 array (LDS or global).
// Lane l: row m=l%16; half=l/16 -> two contiguous 16B runs (two b128 loads).
__device__ __forceinline__ v16h load_afrag(const f16* base, int rowStride,
                                           int k0, int lane) {
  const int m = lane & 15, half = lane >> 4;
  const f16* p = base + m * rowStride + k0 + half * 8;
  v8h lo = *(const v8h*)(p);
  v8h hi = *(const v8h*)(p + 16);
  v16h r;
#pragma unroll
  for (int i = 0; i < 8; ++i) { r[i] = lo[i]; r[i + 8] = hi[i]; }
  return r;
}

// B fragment: prepacked so each lane reads 32 contiguous bytes.
__device__ __forceinline__ v16h load_bfrag(const f16* blk, int lane) {
  const f16* p = blk + lane * 16;
  v8h lo = *(const v8h*)(p);
  v8h hi = *(const v8h*)(p + 8);
  v16h r;
#pragma unroll
  for (int i = 0; i < 8; ++i) { r[i] = lo[i]; r[i + 8] = hi[i]; }
  return r;
}

__device__ __forceinline__ float elu1(float v) { return v > 0.f ? v : expm1f(v); }

// Store a 16x16 f32 accumulator tile (f16-converted) to LDS (current step) and
// global history (future taps).
__device__ __forceinline__ void store_tile(f16* lds, int ldsStride, f16* g,
                                           int gStride, int n0, int lane,
                                           v8f vals) {
  const int n = n0 + (lane & 15), half = lane >> 4;
#pragma unroll
  for (int r = 0; r < 8; ++r) {
    f16 v = (f16)vals[r];
    lds[(r + 8 * half) * ldsStride + n] = v;
    g[(r + 8 * half) * gStride + n] = v;
  }
}

// Causal K=3 dilated conv GEMM for one wave's 16-wide N tile.
// t0/t1 = older-tap tiles (TDM-staged LDS or direct global), cur = current LDS.
__device__ __forceinline__ v8f conv_acc(const f16* t0, const f16* t1,
                                        const f16* cur, int chans,
                                        const f16* W, int t, int d, int nt,
                                        int lane) {
  v8f acc = {};
  const int kcN = chans >> 5;
  const f16* srcs[3] = {t0, t1, cur};
#pragma unroll
  for (int tap = 0; tap < 3; ++tap) {
    if (t - (2 - tap) * d < 0) continue;  // causal zero pad: zeros add nothing
    const f16* a = srcs[tap];
#pragma unroll
    for (int kc = 0; kc < kcN; ++kc) {
      v16h av = load_afrag(a, chans, kc * 32, lane);
      v16h bv = load_bfrag(W + (size_t)((tap * kcN + kc) * 8 + nt) * 512, lane);
      acc = wmma16(av, bv, acc);
    }
  }
  return acc;
}

// ---------------------------------------------------------------------------
// LDS layout (halves) — single symbol so TDM lds_addr offsets start at 0.
// ---------------------------------------------------------------------------
#define OFF_X   0        // 16*64   current x_t
#define OFF_H   1024     // 16*128  current block input/output h
#define OFF_C   3072     // 16*128  current intra-block c
#define OFF_E   5120     // 16*128  enc_last
#define OFF_D0  7168     // 16*128  head layer0 out
#define OFF_D1  9216     // 16*64   head layer1 out
#define OFF_STG0 10240   // blk0 staging: c1 taps 2x1024, c2 taps 2x2048
#define OFF_STG1 16384   // blk1..7 staging: 4x2048 each
#define OFF_PRED 73728   // 16 floats
#define SMEM_HALVES 73760

__device__ __forceinline__ int stg_off(int blk, int which, int tap) {
  // which: 0 = conv1 input taps, 1 = conv2 input taps
  if (blk == 0)
    return which == 0 ? (OFF_STG0 + tap * 1024) : (OFF_STG0 + 2048 + tap * 2048);
  return OFF_STG1 + (blk - 1) * 8192 + which * 4096 + tap * 2048;
}

#if USE_TDM
// Issue one TDM DMA: contiguous `bytes` from global -> LDS offset. D# per
// CDNA5 ISA: count=1, type=2, data_size=8B, 1-row tile.
__device__ __forceinline__ void tdm_load(unsigned lds_bytes, const void* gptr,
                                         unsigned bytes) {
  unsigned long long ga = (unsigned long long)gptr;
  unsigned elems = bytes >> 3;  // 8-byte elements
  u32x4 g0;
  g0[0] = 1u;                                              // count=1, user mode
  g0[1] = lds_bytes;                                       // lds_addr
  g0[2] = (unsigned)(ga & 0xFFFFFFFFu);                    // global_addr lo
  g0[3] = (unsigned)((ga >> 32) & 0x01FFFFFFu) | (2u << 30);  // hi | type=2
  i32x8 g1;
  g1[0] = (3 << 16);                       // data_size=3 (8B), no multicast
  g1[1] = (int)((elems & 0xFFFFu) << 16);  // tensor_dim0[15:0] @ bits 63:48
  g1[2] = (int)((elems >> 16) & 0xFFFFu) | (1 << 16);  // dim0 hi | tensor_dim1=1
  g1[3] = (int)((elems & 0xFFFFu) << 16);  // tile_dim0 @ bits 127:112
  g1[4] = 1;                               // tile_dim1=1
  g1[5] = (int)elems;                      // tensor_dim0_stride lo
  g1[6] = 0;
  g1[7] = 0;
  i32x4 z4 = {0, 0, 0, 0};
#if defined(__clang_major__) && __clang_major__ >= 23
  i32x8 z8 = {0, 0, 0, 0, 0, 0, 0, 0};
  __builtin_amdgcn_tensor_load_to_lds(g0, g1, z4, z4, z8, 0);
#else
  __builtin_amdgcn_tensor_load_to_lds(g0, g1, z4, z4, 0);
#endif
}

// 32 TDM ops per timestep, 4 per block, issued in block order; TENSORcnt
// decrements in order -> constant per-block thresholds.
__device__ __forceinline__ void wait_tensor_const(int blk) {
  switch (blk) {
    case 0: __builtin_amdgcn_s_wait_tensorcnt(28); break;
    case 1: __builtin_amdgcn_s_wait_tensorcnt(24); break;
    case 2: __builtin_amdgcn_s_wait_tensorcnt(20); break;
    case 3: __builtin_amdgcn_s_wait_tensorcnt(16); break;
    case 4: __builtin_amdgcn_s_wait_tensorcnt(12); break;
    case 5: __builtin_amdgcn_s_wait_tensorcnt(8); break;
    case 6: __builtin_amdgcn_s_wait_tensorcnt(4); break;
    default: __builtin_amdgcn_s_wait_tensorcnt(0); break;
  }
}
#endif  // USE_TDM

// ---------------------------------------------------------------------------
// Weight packing: f32 (taps, KinReal, N) -> f16 B-fragment blocks
// [tap][kchunk][ntile][lane][16 halves]; K padded KinReal -> KinPad with zeros.
// ---------------------------------------------------------------------------
__global__ __launch_bounds__(256) void pack_wb_kernel(
    const float* __restrict__ src, f16* __restrict__ dst, int taps, int KinReal,
    int KinPad, int N) {
  const int kcN = KinPad >> 5;
  const int ntiles = N >> 4;
  const long total = (long)taps * kcN * ntiles * 512;
  for (long i = (long)blockIdx.x * blockDim.x + threadIdx.x; i < total;
       i += (long)gridDim.x * blockDim.x) {
    int h = (int)(i & 15);
    long j = i >> 4;
    int lane = (int)(j & 31); j >>= 5;
    int ntile = (int)(j % ntiles); j /= ntiles;
    int kc = (int)(j % kcN);
    int tap = (int)(j / kcN);
    int v = h >> 1, s = h & 1, half = lane >> 4;
    int k = kc * 32 + (v >> 2) * 16 + half * 8 + (v & 3) * 2 + s;
    int n = ntile * 16 + (lane & 15);
    float val = (k < KinReal) ? src[((long)tap * KinReal + k) * N + n] : 0.f;
    dst[i] = (f16)val;
  }
}

// ---------------------------------------------------------------------------
// Persistent decoder: 4 workgroups x 16 batch rows, 8 waves (one N-tile each).
// ---------------------------------------------------------------------------
__global__ __launch_bounds__(256) void tcn_decoder_kernel(
    const float* __restrict__ data_encoder, const float* __restrict__ data_decoder,
    const float* __restrict__ last_y,
    const float* __restrict__ b_in, const float* __restrict__ b_res,
    const float* __restrict__ b1, const float* __restrict__ b2,
    const float* __restrict__ bd0, const float* __restrict__ bd1,
    const float* __restrict__ bd2, const float* __restrict__ wd2,
    const f16* __restrict__ pw_in, const f16* __restrict__ pw_res,
    const f16* __restrict__ pw1, const f16* __restrict__ pw2,
    const f16* __restrict__ pwd0, const f16* __restrict__ pwd1,
    f16* __restrict__ x0hist, f16* __restrict__ chist, f16* __restrict__ hhist,
    float* __restrict__ out) {
  __shared__ __align__(16) f16 smem[SMEM_HALVES];
  f16* sX  = smem + OFF_X;
  f16* sH  = smem + OFF_H;
  f16* sC  = smem + OFF_C;
  f16* sE  = smem + OFF_E;
  f16* sD0 = smem + OFF_D0;
  f16* sD1 = smem + OFF_D1;
  float* sPred = (float*)(smem + OFF_PRED);

  const int wg = blockIdx.x;
  const int tid = threadIdx.x;
  const int lane = tid & 31;
  const int wave = tid >> 5;
  const int b0 = wg * 16;
  const int nt = wave, n0 = nt * 16;

  for (int i = tid; i < 16 * 128; i += 256) {
    int m = i >> 7, c = i & 127;
    sE[i] = (f16)data_encoder[((size_t)(b0 + m) * 128 + 127) * 128 + c];
  }

  const size_t CH = (size_t)32 * 16 * 128;  // per-layer history: [t][m][c]
  f16* X0 = x0hist + (size_t)wg * 32 * 16 * 64;
  f16* C0 = chist + (size_t)wg * 8 * CH;
  f16* H0 = hhist + (size_t)wg * 8 * CH;
  const int dil[8] = {1, 2, 4, 8, 1, 2, 4, 8};
  __syncthreads();

  for (int t = 0; t < 32; ++t) {
#if USE_TDM
    // Prefetch every older-tap tile for this timestep via the Tensor Data
    // Mover (wave 0 issues; EXEC ignored). t<0 taps are issued into unused
    // slots (addresses stay inside d_ws) so wait thresholds stay constant;
    // the GEMMs below still skip them.
    if (wave == 0) {
      tdm_load((unsigned)stg_off(0, 0, 0) * 2, X0 + (long)(t - 2) * 1024, 2048);
      tdm_load((unsigned)stg_off(0, 0, 1) * 2, X0 + (long)(t - 1) * 1024, 2048);
      tdm_load((unsigned)stg_off(0, 1, 0) * 2, C0 + (long)(t - 2) * 2048, 4096);
      tdm_load((unsigned)stg_off(0, 1, 1) * 2, C0 + (long)(t - 1) * 2048, 4096);
      for (int blk = 1; blk < 8; ++blk) {
        int d = dil[blk];
        const f16* hsrc = H0 + (size_t)(blk - 1) * CH;
        const f16* csrc = C0 + (size_t)blk * CH;
        tdm_load((unsigned)stg_off(blk, 0, 0) * 2, hsrc + (long)(t - 2 * d) * 2048, 4096);
        tdm_load((unsigned)stg_off(blk, 0, 1) * 2, hsrc + (long)(t - d) * 2048, 4096);
        tdm_load((unsigned)stg_off(blk, 1, 0) * 2, csrc + (long)(t - 2 * d) * 2048, 4096);
        tdm_load((unsigned)stg_off(blk, 1, 1) * 2, csrc + (long)(t - d) * 2048, 4096);
      }
    }
#endif
    // ---- build x_t: decoder features + autoregressive y, zero pad to 64 ----
    for (int i = tid; i < 16 * 64; i += 256) {
      int m = i >> 6, c = i & 63;
      float v = 0.f;
      if (c < 32) v = data_decoder[((size_t)(b0 + m) * 32 + t) * 32 + c];
      else if (c == 32) v = (t == 0) ? last_y[b0 + m] : sPred[m];
      f16 h = (f16)v;
      sX[i] = h;
      X0[(size_t)(t * 16 + m) * 64 + c] = h;
    }
    __syncthreads();

    for (int blk = 0; blk < 8; ++blk) {
      const int d = dil[blk];
#if USE_TDM
      if (wave == 0) wait_tensor_const(blk);  // this block's taps are in LDS
      __syncthreads();
#endif
      // ---- conv1 (x->F for blk 0, F->F otherwise) ----
      v8f acc;
      if (blk == 0) {
#if USE_TDM
        acc = conv_acc(smem + stg_off(0, 0, 0), smem + stg_off(0, 0, 1), sX, 64,
                       pw_in, t, 1, nt, lane);
#else
        acc = conv_acc(X0 + (long)(t - 2) * 1024, X0 + (long)(t - 1) * 1024, sX,
                       64, pw_in, t, 1, nt, lane);
#endif
      } else {
        const f16* W = pw1 + (size_t)(blk - 1) * 49152;
#if USE_TDM
        acc = conv_acc(smem + stg_off(blk, 0, 0), smem + stg_off(blk, 0, 1), sH,
                       128, W, t, d, nt, lane);
#else
        const f16* hh = H0 + (size_t)(blk - 1) * CH;
        acc = conv_acc(hh + (long)(t - 2 * d) * 2048, hh + (long)(t - d) * 2048,
                       sH, 128, W, t, d, nt, lane);
#endif
      }
      {
        const float* bias = (blk == 0) ? b_in : (b1 + (blk - 1) * 128);
        float bv = bias[n0 + (lane & 15)];
        v8f vals;
#pragma unroll
        for (int r = 0; r < 8; ++r) vals[r] = elu1(acc[r] + bv);
        store_tile(sC, 128, C0 + (size_t)blk * CH + (size_t)t * 16 * 128, 128,
                   n0, lane, vals);
      }
      // ---- residual (1x1 conv of x for blk 0, identity otherwise) ----
      v8f res;
      if (blk == 0) {
        v8f r = {};
#pragma unroll
        for (int kc = 0; kc < 2; ++kc) {
          v16h a = load_afrag(sX, 64, kc * 32, lane);
          v16h b = load_bfrag(pw_res + (size_t)(kc * 8 + nt) * 512, lane);
          r = wmma16(a, b, r);
        }
        float bv = b_res[n0 + (lane & 15)];
#pragma unroll
        for (int i = 0; i < 8; ++i) res[i] = r[i] + bv;
      } else {
        int n = n0 + (lane & 15), half = lane >> 4;
#pragma unroll
        for (int r = 0; r < 8; ++r) res[r] = (float)sH[(r + 8 * half) * 128 + n];
      }
      __syncthreads();  // sC visible to all waves; res reads of sH done
      // ---- conv2 (F->F) + residual + elu ----
      v8f acc2;
      {
        const f16* W2 = pw2 + (size_t)blk * 49152;
#if USE_TDM
        acc2 = conv_acc(smem + stg_off(blk, 1, 0), smem + stg_off(blk, 1, 1),
                        sC, 128, W2, t, d, nt, lane);
#else
        const f16* cc = C0 + (size_t)blk * CH;
        acc2 = conv_acc(cc + (long)(t - 2 * d) * 2048, cc + (long)(t - d) * 2048,
                        sC, 128, W2, t, d, nt, lane);
#endif
      }
      {
        float bv = b2[blk * 128 + n0 + (lane & 15)];
        v8f vals;
#pragma unroll
        for (int r = 0; r < 8; ++r) {
          float c2 = elu1(acc2[r] + bv);
          vals[r] = elu1(c2 + res[r]);
        }
        store_tile(sH, 128, H0 + (size_t)blk * CH + (size_t)t * 16 * 128, 128,
                   n0, lane, vals);
      }
      __syncthreads();
    }

    // ---- head layer 0: concat(h,enc) 256 -> 128, elu ----
    v8f a0 = {};
#pragma unroll
    for (int kc = 0; kc < 4; ++kc) {
      v16h a = load_afrag(sH, 128, kc * 32, lane);
      v16h b = load_bfrag(pwd0 + (size_t)(kc * 8 + nt) * 512, lane);
      a0 = wmma16(a, b, a0);
    }
#pragma unroll
    for (int kc = 0; kc < 4; ++kc) {
      v16h a = load_afrag(sE, 128, kc * 32, lane);
      v16h b = load_bfrag(pwd0 + (size_t)((4 + kc) * 8 + nt) * 512, lane);
      a0 = wmma16(a, b, a0);
    }
    {
      int n = n0 + (lane & 15), half = lane >> 4;
      float bv = bd0[n];
#pragma unroll
      for (int r = 0; r < 8; ++r)
        sD0[(r + 8 * half) * 128 + n] = (f16)elu1(a0[r] + bv);
    }
    __syncthreads();
    // ---- head layer 1: 128 -> 64, elu (waves 0..3, full EXEC per wave) ----
    if (wave < 4) {
      v8f a1 = {};
#pragma unroll
      for (int kc = 0; kc < 4; ++kc) {
        v16h a = load_afrag(sD0, 128, kc * 32, lane);
        v16h b = load_bfrag(pwd1 + (size_t)(kc * 4 + wave) * 512, lane);
        a1 = wmma16(a, b, a1);
      }
      int n = wave * 16 + (lane & 15), half = lane >> 4;
      float bv = bd1[n];
#pragma unroll
      for (int r = 0; r < 8; ++r)
        sD1[(r + 8 * half) * 64 + n] = (f16)elu1(a1[r] + bv);
    }
    __syncthreads();
    // ---- head layer 2: 64 -> 1 ----
    if (tid < 16) {
      float s = bd2[0];
      for (int c = 0; c < 64; ++c) s += (float)sD1[tid * 64 + c] * wd2[c];
      sPred[tid] = s;
      out[(size_t)(b0 + tid) * 32 + t] = s;
    }
    __threadfence();  // history of step t visible (vector + TDM) to later steps
    __syncthreads();
  }
}

// ---------------------------------------------------------------------------
// Host launcher
// ---------------------------------------------------------------------------
extern "C" void kernel_launch(void* const* d_in, const int* in_sizes, int n_in,
                              void* d_out, int out_size, void* d_ws,
                              size_t ws_size, hipStream_t stream) {
  const float* data_encoder = (const float*)d_in[0];
  const float* data_decoder = (const float*)d_in[1];
  const float* last_y = (const float*)d_in[2];
  const float* w_in  = (const float*)d_in[3];
  const float* b_in  = (const float*)d_in[4];
  const float* w_res = (const float*)d_in[5];
  const float* b_res = (const float*)d_in[6];
  const float* w1  = (const float*)d_in[7];
  const float* b1  = (const float*)d_in[8];
  const float* w2  = (const float*)d_in[9];
  const float* b2  = (const float*)d_in[10];
  const float* wd0 = (const float*)d_in[11];
  const float* bd0 = (const float*)d_in[12];
  const float* wd1 = (const float*)d_in[13];
  const float* bd1 = (const float*)d_in[14];
  const float* wd2 = (const float*)d_in[15];
  const float* bd2 = (const float*)d_in[16];
  (void)in_sizes; (void)n_in; (void)out_size; (void)ws_size;

  char* ws = (char*)d_ws;
  size_t off = 0;
  auto carve = [&](size_t halves) {
    size_t o = off;
    off += (halves * 2 + 255) & ~(size_t)255;
    return o;
  };
  size_t o_pwin  = carve((size_t)3 * 2 * 8 * 512);       // w_in  packed
  size_t o_pwres = carve((size_t)2 * 8 * 512);           // w_res packed
  size_t o_pw1   = carve((size_t)21 * 4 * 8 * 512);      // w1 packed (7 layers)
  size_t o_pw2   = carve((size_t)24 * 4 * 8 * 512);      // w2 packed (8 layers)
  size_t o_pwd0  = carve((size_t)8 * 8 * 512);           // wd0 packed
  size_t o_pwd1  = carve((size_t)4 * 4 * 512);           // wd1 packed
  size_t o_x0    = carve((size_t)4 * 32 * 16 * 64);      // x history
  size_t o_ch    = carve((size_t)4 * 8 * 32 * 16 * 128); // c history
  size_t o_hh    = carve((size_t)4 * 8 * 32 * 16 * 128); // h history

  f16* pw_in  = (f16*)(ws + o_pwin);
  f16* pw_res = (f16*)(ws + o_pwres);
  f16* pw1    = (f16*)(ws + o_pw1);
  f16* pw2    = (f16*)(ws + o_pw2);
  f16* pwd0   = (f16*)(ws + o_pwd0);
  f16* pwd1   = (f16*)(ws + o_pwd1);
  f16* x0hist = (f16*)(ws + o_x0);
  f16* chist  = (f16*)(ws + o_ch);
  f16* hhist  = (f16*)(ws + o_hh);

  // Pack weights into WMMA B-fragment layout (f32 -> f16, K padded).
  pack_wb_kernel<<<96,  256, 0, stream>>>(w_in,  pw_in,  3, 33, 64, 128);
  pack_wb_kernel<<<32,  256, 0, stream>>>(w_res, pw_res, 1, 33, 64, 128);
  pack_wb_kernel<<<512, 256, 0, stream>>>(w1,  pw1, 21, 128, 128, 128);
  pack_wb_kernel<<<512, 256, 0, stream>>>(w2,  pw2, 24, 128, 128, 128);
  pack_wb_kernel<<<128, 256, 0, stream>>>(wd0, pwd0, 1, 256, 256, 128);
  pack_wb_kernel<<<32,  256, 0, stream>>>(wd1, pwd1, 1, 128, 128, 64);

  // Persistent decode: 4 workgroups x (16 batch rows, 8 waves).
  tcn_decoder_kernel<<<4, 256, 0, stream>>>(
      data_encoder, data_decoder, last_y, b_in, b_res, b1, b2, bd0, bd1, bd2,
      wd2, pw_in, pw_res, pw1, pw2, pwd0, pwd1, x0hist, chist, hhist,
      (float*)d_out);
}
